// HighResFCGQCNN_22866405884156
// MI455X (gfx1250) — compile-verified, matrix operand order checked by reading, not run on hardware
//
#include <hip/hip_runtime.h>
#include <hip/hip_bf16.h>

typedef __attribute__((ext_vector_type(16))) _Float16 v16h;
typedef __attribute__((ext_vector_type(8)))  _Float16 v8h;
typedef __attribute__((ext_vector_type(8)))  float    v8f;

// ---------------------------------------------------------------------------
// Small helper kernels
// ---------------------------------------------------------------------------

__global__ void pack_f16_kernel(const float* __restrict__ src,
                                _Float16* __restrict__ dst, int n) {
  int i = blockIdx.x * blockDim.x + threadIdx.x;
  if (i < n) dst[i] = (_Float16)src[i];
}

// conv1: (2,1,128,128) -> (2,64,128,128), 7x7 SAME + bias + relu
__global__ void conv1_kernel(const float* __restrict__ x,
                             const float* __restrict__ w,
                             const float* __restrict__ b,
                             float* __restrict__ out) {
  int t = blockIdx.x * blockDim.x + threadIdx.x;       // 2*64*128*128
  int xx = t & 127;
  int yy = (t >> 7) & 127;
  int o  = (t >> 14) & 63;
  int bb = t >> 20;
  const float* xin = x + (size_t)bb * 128 * 128;
  const float* wo  = w + o * 49;
  float s = b[o];
#pragma unroll
  for (int ky = 0; ky < 7; ++ky) {
    int iy = yy + ky - 3;
    if (iy < 0 || iy > 127) continue;
#pragma unroll
    for (int kx = 0; kx < 7; ++kx) {
      int ix = xx + kx - 3;
      if (ix < 0 || ix > 127) continue;
      s += xin[iy * 128 + ix] * wo[ky * 7 + kx];
    }
  }
  out[t] = fmaxf(s, 0.f);
}

__device__ __forceinline__ float lrn_val(float cur, float prev) {
  // size=2, alpha=2e-5, beta=0.75, k=1:  x / (1 + a*0.5*(x^2 + xprev^2))^0.75
  float avg = 0.5f * (cur * cur + prev * prev);
  return cur * __powf(1.f + 2e-5f * avg, -0.75f);
}

// LRN on h2 (2,64,128,128) fused with the 4 shifted crops + 2x2 maxpool.
// Output: (8,64,63,63); group g uses crop offset (dy=g&1, dx=g>>1).
__global__ void lrn_crop_pool_kernel(const float* __restrict__ h,
                                     float* __restrict__ out) {
  int t = blockIdx.x * blockDim.x + threadIdx.x;       // 8*64*63*63
  int px = t % 63;
  int py = (t / 63) % 63;
  int c  = (t / (63 * 63)) % 64;
  int gb = t / (63 * 63 * 64);
  int gi = gb >> 1, bb = gb & 1;
  int dy = gi & 1, dx = gi >> 1;
  const float* cc = h + ((size_t)(bb * 64 + c) * 128) * 128;
  const float* cp = cc - 128 * 128;
  float m = -3.4e38f;
#pragma unroll
  for (int u = 0; u < 2; ++u)
#pragma unroll
    for (int v = 0; v < 2; ++v) {
      int y = dy + 2 * py + u;
      int x = dx + 2 * px + v;
      float cur  = cc[y * 128 + x];
      float prev = (c > 0) ? cp[y * 128 + x] : 0.f;
      m = fmaxf(m, lrn_val(cur, prev));
    }
  out[t] = m;
}

// Elementwise LRN on (8,64,63,63)
__global__ void lrn_kernel(const float* __restrict__ h, float* __restrict__ out) {
  int t = blockIdx.x * blockDim.x + threadIdx.x;       // 8*64*63*63
  int c = (t / (63 * 63)) % 64;
  float cur  = h[t];
  float prev = (c > 0) ? h[t - 63 * 63] : 0.f;
  out[t] = lrn_val(cur, prev);
}

// conv7 (1x1, 1024->1) + sigmoid + interweave of the 4 crop groups.
// h6: (8,1024,48,48) f16  ->  out: (2,1,96,96) f32
__global__ void conv7_sigmoid_interweave_kernel(const _Float16* __restrict__ h6,
                                                const float* __restrict__ w7,
                                                const float* __restrict__ b7,
                                                float* __restrict__ out) {
  int t = blockIdx.x * blockDim.x + threadIdx.x;       // 2*96*96
  int X  = t % 96;
  int Y  = (t / 96) % 96;
  int bb = t / (96 * 96);
  int j = Y & 1, i = X & 1, hh = Y >> 1, ww = X >> 1;
  int g  = (i << 1) | j;                               // i1,i2,i3,i4 selection
  int gb = g * 2 + bb;
  const _Float16* p = h6 + (size_t)gb * 1024 * 2304 + hh * 48 + ww;
  float s = b7[0];
  for (int ch = 0; ch < 1024; ++ch) s += (float)p[(size_t)ch * 2304] * w7[ch];
  out[t] = 1.f / (1.f + __expf(-s));
}

// ---------------------------------------------------------------------------
// Implicit-im2col GEMM convolution with WMMA f16 (f32 accumulate).
//   out[b,m,oy,ox] = relu( bias[m] + sum_{c,r,s} in[b,c,oy+r-PAD,ox+s-PAD] * w[m,c,r,s] )
// A = packed f16 weights [M][K], K = Cin*R*S (OIHW flatten == K-major).
// Block: 256 threads (8 waves), tile BM x 128, K-stage 64 (two WMMA slabs).
// Double-buffered LDS, software-pipelined: one barrier per 2*MT*NT WMMAs.
// Waves: 2 (M) x 4 (N); each wave computes (BM/2) x 32 via 16x16 WMMA tiles.
// ---------------------------------------------------------------------------
template <int BM, int Bn, int Cin, int H, int W, int M, int R, int S, int PAD,
          int OH, int OW, bool RELU, bool IN_F16, bool OUT_F16>
__global__ void __launch_bounds__(256)
gemm_conv_wmma(const void* __restrict__ inp, const _Float16* __restrict__ wpk,
               const float* __restrict__ bias, void* __restrict__ outp) {
  constexpr int K      = Cin * R * S;     // multiple of 64 for all instantiations
  constexpr int KSTEPS = K / 64;
  constexpr int NPIX   = OH * OW;
  constexpr int Ntot   = Bn * NPIX;
  constexpr int BN     = 128;
  constexpr int LDK    = 72;              // 64 + 8 halves pad (conflict-free b128)
  constexpr int WM     = BM / 2;
  constexpr int MT     = WM / 16;         // 2 (BM=64) or 4 (BM=128)
  constexpr int NT     = 2;               // 32/16
  constexpr int ACH    = BM / 32;         // A row-chunks per thread

  __shared__ _Float16 As[2][BM][LDK];     // [buf][m][k]
  __shared__ _Float16 Bs[2][BN][LDK];     // [buf][n][k]  (transposed)

  const int t    = threadIdx.x;
  const int lane = t & 31;
  const int wid  = t >> 5;
  const int wm   = wid & 1;               // wave M index (0..1)
  const int wn   = wid >> 1;              // wave N index (0..3)
  const int l16  = lane & 15;
  const int half = lane >> 4;
  const int kb   = half * 8;              // K-octet select per half-wave

  const int mBlock = blockIdx.y * BM;
  const int nBlock = blockIdx.x * BN;

  // B-stage mapping: thread owns k-rows {bk, bk+32} x 16 consecutive n's.
  const int bk  = t >> 3;                 // 0..31
  const int bn0 = (t & 7) * 16;           // 0..112
  // A-stage mapping: 8 contiguous halves along K, rows arow + c*32.
  const int arow  = t >> 3;               // 0..31
  const int akoff = (t & 7) * 8;          // 0..56

  // ---- hoisted im2col n-decomposition (K-loop invariant) -------------------
  // idx(b,c,oy+dy,ox+dx) = base_j + (c*H*W + dy*W + dx), offset thread-uniform.
  const int jmax = Ntot - (nBlock + bn0); // element valid iff j < jmax
  int basev[16], oyv[16], oxv[16];
#pragma unroll
  for (int j = 0; j < 16; ++j) {
    int n = nBlock + bn0 + j;
    if (n > Ntot - 1) n = Ntot - 1;       // clamp: decomposition stays in range
    const int b   = n / NPIX;
    const int rem = n % NPIX;
    oyv[j]   = rem / OW;
    oxv[j]   = rem % OW;
    basev[j] = ((b * Cin) * H + oyv[j]) * W + oxv[j];
  }

  v8f acc[MT][NT];
  v8f zacc = {};
#pragma unroll
  for (int i = 0; i < MT; ++i)
#pragma unroll
    for (int j = 0; j < NT; ++j) acc[i][j] = zacc;

  v8h     areg[ACH];
  _Float16 breg[2][16];

  // ---- pipeline stages -----------------------------------------------------
  auto global_load = [&](int ks) {
    const int k0 = ks * 64;
#pragma unroll
    for (int c = 0; c < ACH; ++c)
      areg[c] =
          *(const v8h*)&wpk[(size_t)(mBlock + arow + c * 32) * K + k0 + akoff];
#pragma unroll
    for (int h2 = 0; h2 < 2; ++h2) {
      const int kg  = k0 + bk + 32 * h2;
      const int cch = kg / (R * S);
      const int rs  = kg % (R * S);
      const int dy  = rs / S - PAD;
      const int dx  = rs % S - PAD;
      const int kof = cch * (H * W) + dy * W + dx;
#pragma unroll
      for (int j = 0; j < 16; ++j) {
        const int iy = oyv[j] + dy;
        const int ix = oxv[j] + dx;
        const bool ok = (j < jmax) && ((unsigned)iy < (unsigned)H) &&
                        ((unsigned)ix < (unsigned)W);
        float v = 0.f;
        if (ok)
          v = IN_F16 ? (float)((const _Float16*)inp)[basev[j] + kof]
                     : ((const float*)inp)[basev[j] + kof];
        breg[h2][j] = (_Float16)v;
      }
    }
  };

  auto lds_store = [&](int buf) {
#pragma unroll
    for (int c = 0; c < ACH; ++c)
      *(v8h*)&As[buf][arow + c * 32][akoff] = areg[c];
#pragma unroll
    for (int h2 = 0; h2 < 2; ++h2)
#pragma unroll
      for (int j = 0; j < 16; ++j)
        Bs[buf][bn0 + j][bk + 32 * h2] = breg[h2][j];
  };

  auto compute = [&](int buf) {
#pragma unroll
    for (int slab = 0; slab < 2; ++slab) {
      const int kk0 = slab * 32;
      v16h afrag[MT], bfrag[NT];
#pragma unroll
      for (int i = 0; i < MT; ++i) {
        const int row = wm * WM + i * 16 + l16;
        v8h lo = *(const v8h*)&As[buf][row][kk0 + kb];       // K = kb..kb+7
        v8h hi = *(const v8h*)&As[buf][row][kk0 + 16 + kb];  // K = 16+kb..
        afrag[i] = __builtin_shufflevector(lo, hi, 0, 1, 2, 3, 4, 5, 6, 7,
                                           8, 9, 10, 11, 12, 13, 14, 15);
      }
#pragma unroll
      for (int j = 0; j < NT; ++j) {
        const int col = wn * 32 + j * 16 + l16;
        v8h lo = *(const v8h*)&Bs[buf][col][kk0 + kb];
        v8h hi = *(const v8h*)&Bs[buf][col][kk0 + 16 + kb];
        bfrag[j] = __builtin_shufflevector(lo, hi, 0, 1, 2, 3, 4, 5, 6, 7,
                                           8, 9, 10, 11, 12, 13, 14, 15);
      }
#pragma unroll
      for (int i = 0; i < MT; ++i)
#pragma unroll
        for (int j = 0; j < NT; ++j)
          acc[i][j] = __builtin_amdgcn_wmma_f32_16x16x32_f16(
              false, afrag[i], false, bfrag[j], (short)0, acc[i][j], false,
              false);
    }
  };

  // ---- pipelined main loop: 1 barrier per K-stage --------------------------
  global_load(0);
  lds_store(0);
  __syncthreads();
  for (int ks = 0; ks < KSTEPS; ++ks) {
    const int  cur  = ks & 1;
    const bool more = (ks + 1 < KSTEPS);
    if (more) global_load(ks + 1);        // vmem issues early ...
    if (ks + 2 < KSTEPS)                  // ... plus L2 prefetch one ahead
      __builtin_prefetch(
          &wpk[(size_t)(mBlock + arow) * K + (size_t)(ks + 2) * 64 + akoff], 0,
          1);
    compute(cur);                         // ... hidden behind WMMA
    if (more) lds_store(cur ^ 1);         // other buffer: no conflict w/ readers
    __syncthreads();
  }

  // ---- epilogue: C/D layout (lane -> N, VGPR v + 8*half -> M); bias+relu
#pragma unroll
  for (int i = 0; i < MT; ++i) {
    const int mbase = mBlock + wm * WM + i * 16 + half * 8;
#pragma unroll
    for (int j = 0; j < NT; ++j) {
      const int n = nBlock + wn * 32 + j * 16 + l16;
      if (n < Ntot) {
        const int b   = n / NPIX;
        const int rem = n % NPIX;
        const int oy  = rem / OW;
        const int ox  = rem % OW;
#pragma unroll
        for (int v = 0; v < 8; ++v) {
          const int mg = mbase + v;
          float val = acc[i][j][v] + bias[mg];
          if (RELU) val = fmaxf(val, 0.f);
          const size_t oidx = (((size_t)b * M + mg) * OH + oy) * OW + ox;
          if (OUT_F16) ((_Float16*)outp)[oidx] = (_Float16)val;
          else         ((float*)outp)[oidx] = val;
        }
      }
    }
  }
}

// ---------------------------------------------------------------------------
// Orchestration
// ---------------------------------------------------------------------------
extern "C" void kernel_launch(void* const* d_in, const int* in_sizes, int n_in,
                              void* d_out, int out_size, void* d_ws, size_t ws_size,
                              hipStream_t stream) {
  (void)in_sizes; (void)n_in; (void)out_size; (void)ws_size;
  const float* x  = (const float*)d_in[0];
  const float* w1 = (const float*)d_in[1];
  const float* b1 = (const float*)d_in[2];
  const float* w2 = (const float*)d_in[3];
  const float* b2 = (const float*)d_in[4];
  const float* w3 = (const float*)d_in[5];
  const float* b3 = (const float*)d_in[6];
  const float* w4 = (const float*)d_in[7];
  const float* b4 = (const float*)d_in[8];
  const float* w5 = (const float*)d_in[9];
  const float* b5 = (const float*)d_in[10];
  const float* w6 = (const float*)d_in[11];
  const float* b6 = (const float*)d_in[12];
  const float* w7 = (const float*)d_in[13];
  const float* b7 = (const float*)d_in[14];

  char* ws = (char*)d_ws;
  float*    h1   = (float*)(ws + 0);            // (2,64,128,128)   8.39 MB
  float*    h2   = (float*)(ws + 8388608);      // (2,64,128,128)   8.39 MB
  float*    pool = (float*)(ws + 16777216);     // (8,64,63,63)     8.13 MB
  float*    h3   = (float*)(ws + 24905728);     // (8,64,63,63)
  float*    h4   = (float*)(ws + 33034240);     // (8,64,63,63)
  float*    lr2  = (float*)(ws + 41162752);     // (8,64,63,63)
  _Float16* wpk  = (_Float16*)(ws + 49291264);  // packed weights  33.55 MB max
  _Float16* h5   = (_Float16*)(ws + 82845696);  // (8,1024,48,48)  37.75 MB
  _Float16* h6   = (_Float16*)(ws + 120594432); // (8,1024,48,48)  37.75 MB
  // total workspace use: 158,343,168 bytes

  dim3 blk(256);

  // conv1 7x7 SAME + relu (direct f32, 0.2 GF)
  conv1_kernel<<<8192, blk, 0, stream>>>(x, w1, b1, h1);

  // conv2 5x5 SAME + relu : M=64, K=1600, N=32768
  pack_f16_kernel<<<400, blk, 0, stream>>>(w2, wpk, 64 * 64 * 25);
  gemm_conv_wmma<64, 2, 64, 128, 128, 64, 5, 5, 2, 128, 128, true, false, false>
      <<<dim3(256, 1), blk, 0, stream>>>(h1, wpk, b2, h2);

  // LRN + 4 shifted crops + 2x2 maxpool -> (8,64,63,63)
  lrn_crop_pool_kernel<<<7938, blk, 0, stream>>>(h2, pool);

  // conv3 3x3 SAME + relu : M=64, K=576, N=31752
  pack_f16_kernel<<<144, blk, 0, stream>>>(w3, wpk, 64 * 64 * 9);
  gemm_conv_wmma<64, 8, 64, 63, 63, 64, 3, 3, 1, 63, 63, true, false, false>
      <<<dim3(249, 1), blk, 0, stream>>>(pool, wpk, b3, h3);

  // conv4 3x3 SAME + relu
  pack_f16_kernel<<<144, blk, 0, stream>>>(w4, wpk, 64 * 64 * 9);
  gemm_conv_wmma<64, 8, 64, 63, 63, 64, 3, 3, 1, 63, 63, true, false, false>
      <<<dim3(249, 1), blk, 0, stream>>>(h3, wpk, b4, h4);

  // LRN2
  lrn_kernel<<<7938, blk, 0, stream>>>(h4, lr2);

  // conv5 16x16 VALID + relu : M=1024, K=16384, N=18432  (the 619 GF GEMM)
  pack_f16_kernel<<<65536, blk, 0, stream>>>(w5, wpk, 1024 * 64 * 256);
  gemm_conv_wmma<128, 8, 64, 63, 63, 1024, 16, 16, 0, 48, 48, true, false, true>
      <<<dim3(144, 8), blk, 0, stream>>>(lr2, wpk, b5, h5);

  // conv6 1x1 + relu : M=1024, K=1024, N=18432 (f16 in, f16 out)
  pack_f16_kernel<<<4096, blk, 0, stream>>>(w6, wpk, 1024 * 1024);
  gemm_conv_wmma<128, 8, 1024, 48, 48, 1024, 1, 1, 0, 48, 48, true, true, true>
      <<<dim3(144, 8), blk, 0, stream>>>(h5, wpk, b6, h6);

  // conv7 1x1 -> sigmoid -> interweave to (2,1,96,96)
  conv7_sigmoid_interweave_kernel<<<72, blk, 0, stream>>>(h6, w7, b7,
                                                          (float*)d_out);
}
